// SwinTransformerBlock_30580167147753
// MI455X (gfx1250) — matrix operand order repaired and checked by reading, not compiled, not run on hardware
//
#include <hip/hip_runtime.h>
#include <hip/hip_bf16.h>
#include <math.h>
#include <stdint.h>

// ---------------------------------------------------------------------------
// Swin Transformer block for MI455X (gfx1250, wave32, WMMA bf16 16x16x32,
// TDM double-buffered weight staging).
// ---------------------------------------------------------------------------

typedef __bf16 bf16;
typedef __attribute__((ext_vector_type(16))) __bf16 v16bf;
typedef __attribute__((ext_vector_type(8)))  __bf16 v8bf;
typedef __attribute__((ext_vector_type(8)))  float  v8f;
typedef __attribute__((ext_vector_type(4)))  unsigned int u32x4;
typedef __attribute__((ext_vector_type(8)))  int i32x8;

static __device__ __forceinline__ v16bf frag_cat(v8bf lo, v8bf hi) {
  return __builtin_shufflevector(lo, hi, 0,1,2,3,4,5,6,7,8,9,10,11,12,13,14,15);
}

#define WMMA_BF16(Af, Bf, Cf) \
  __builtin_amdgcn_wmma_f32_16x16x32_bf16(false, (Af), false, (Bf), (short)0, (Cf), false, false)

// ---- problem constants ----
constexpr int BATCH  = 16;
constexpr int HH     = 56;                 // H == W
constexpr int C      = 384;
constexpr int HEADS  = 12;
constexpr int WSZ    = 7;
constexpr int SHIFT  = 3;
constexpr int NTOK   = 49;                 // tokens per window
constexpr int NWIN   = 64;                 // windows per image
constexpr int HID    = 1536;
constexpr int HD     = 32;                 // head dim
constexpr int TOKENS = BATCH * HH * HH;    // 50176 (== B_ * NTOK)
constexpr float SCALE = 0.17677669529663687f;  // 32^-0.5
constexpr float EPS   = 1e-5f;

// ---------------------------------------------------------------------------
// TDM: async DMA of a 2-D bf16 tile [64 rows x 32 elems], row stride
// `strideElems`, from global memory into LDS at byte offset `lds_off`.
// Uses the 2-SGPR-group descriptor form (tile <= 2D, VADDR2/3 = NULL).
// Tracked on TENSORcnt; completion via s_wait_tensorcnt.
// ---------------------------------------------------------------------------
static __device__ __forceinline__ void tdm_load_tile64x32(unsigned lds_off,
                                                          const void* gptr,
                                                          unsigned strideElems) {
  const unsigned long long ga = (unsigned long long)(uintptr_t)gptr;
  u32x4 g0;
  g0[0] = 1u;                                         // count=1, user D#
  g0[1] = lds_off;                                    // lds_addr (bytes)
  g0[2] = (unsigned)(ga & 0xffffffffu);               // global_addr[31:0]
  g0[3] = (unsigned)((ga >> 32) & 0x01ffffffu)        // global_addr[56:32]
          | (2u << 30);                               // type=2 ("image")
  const unsigned td0 = 1u << 30, td1 = 1u << 20;      // huge dims: no OOB clip
  i32x8 g1;
  g1[0] = (int)(1u << 16);                            // data_size=1 (2 bytes)
  g1[1] = (int)((td0 & 0xffffu) << 16);               // tensor_dim0[15:0]
  g1[2] = (int)((td0 >> 16) | ((td1 & 0xffffu) << 16));
  g1[3] = (int)((td1 >> 16) | (32u << 16));           // tile_dim0 = 32
  g1[4] = (int)64u;                                   // tile_dim1 = 64, tile_dim2 = 0
  g1[5] = (int)strideElems;                           // tensor_dim0_stride[31:0]
  g1[6] = 0;                                          // stride[47:32], dim1_stride lo
  g1[7] = 0;
  asm volatile("tensor_load_to_lds %0, %1" :: "s"(g0), "s"(g1) : "memory");
}

// ---------------------------------------------------------------------------
// fp32 -> bf16 weight conversion
// ---------------------------------------------------------------------------
__global__ void cvt_kernel(const float* __restrict__ in, bf16* __restrict__ out, int n) {
  int i = blockIdx.x * 256 + threadIdx.x;
  if (i < n) out[i] = (bf16)in[i];
}

// ---------------------------------------------------------------------------
// LayerNorm.  windowed==1: also apply cyclic shift + window partition and
// write rows in (batch*window, token) order for the QKV GEMM.
// One wave per token (C=384 -> 12 elements/lane), 8 tokens per 256-thread block.
// ---------------------------------------------------------------------------
__global__ void ln_kernel(const float* __restrict__ x,
                          const float* __restrict__ g,
                          const float* __restrict__ b,
                          bf16* __restrict__ out, int windowed) {
  const int wave = threadIdx.x >> 5;
  const int lane = threadIdx.x & 31;
  const int tokIdx = blockIdx.x * 8 + wave;          // [0, 50176)
  const float* row = x + (size_t)tokIdx * C;

  float v[12];
  float s = 0.f;
#pragma unroll
  for (int i = 0; i < 12; ++i) { v[i] = row[lane + i * 32]; s += v[i]; }
#pragma unroll
  for (int o = 16; o > 0; o >>= 1) s += __shfl_xor(s, o, 32);
  const float mean = s * (1.0f / C);

  float var = 0.f;
#pragma unroll
  for (int i = 0; i < 12; ++i) { float d = v[i] - mean; var += d * d; }
#pragma unroll
  for (int o = 16; o > 0; o >>= 1) var += __shfl_xor(var, o, 32);
  const float inv = rsqrtf(var * (1.0f / C) + EPS);

  size_t dstRow;
  if (windowed) {
    const int bi = tokIdx / (HH * HH);
    const int sp = tokIdx - bi * HH * HH;
    const int h  = sp / HH, w = sp - (sp / HH) * HH;
    const int hh = (h + HH - SHIFT) % HH;            // roll(-SHIFT)
    const int ww = (w + HH - SHIFT) % HH;
    const int win = (hh / WSZ) * (HH / WSZ) + (ww / WSZ);
    const int tok = (hh % WSZ) * WSZ + (ww % WSZ);
    dstRow = (size_t)(bi * NWIN + win) * NTOK + tok;
  } else {
    dstRow = (size_t)tokIdx;
  }

  bf16* dp = out + dstRow * C;
#pragma unroll
  for (int i = 0; i < 12; ++i) {
    const int c = lane + i * 32;
    dp[c] = (bf16)((v[i] - mean) * inv * g[c] + b[c]);
  }
}

// ---------------------------------------------------------------------------
// Generic bf16 WMMA GEMM:  out = A[M,K] @ W[N,K]^T + bias, with epilogue.
// 128 threads = 4 waves; block tile 128(M) x 64(N); wave tile 32 x 64
// (2 A-fragments x 4 B-fragments = 8 WMMA per 32-deep k-step).
// B tiles DMA'd into double-buffered LDS by wave 0 via tensor_load_to_lds,
// one k-step ahead of compute; A fragments load direct from global.
// ---------------------------------------------------------------------------
enum { EPI_QKV = 0, EPI_PROJ = 1, EPI_FC1 = 2, EPI_FC2 = 3 };

template <int EPI>
__global__ void gemm_wmma(const bf16* __restrict__ A, const bf16* __restrict__ W,
                          const float* __restrict__ bias,
                          void* __restrict__ outp,
                          const float* __restrict__ resid,
                          int M, int N, int K) {
  __shared__ __align__(16) bf16 Bs[2][64 * 32];

  const int tid  = threadIdx.x;
  const int wave = tid >> 5;
  const int lane = tid & 31;
  const int half = lane >> 4;     // 0/1 : K-half select for WMMA fragments
  const int l15  = lane & 15;
  const int nBase = blockIdx.x * 64;
  const int mBase = blockIdx.y * 128;

  v8f acc[2][4];
#pragma unroll
  for (int p = 0; p < 2; ++p)
#pragma unroll
    for (int t = 0; t < 4; ++t)
#pragma unroll
      for (int i = 0; i < 8; ++i) acc[p][t][i] = 0.f;

  const bf16* aRow0 = A + (size_t)(mBase + wave * 32 + l15) * K;
  const bf16* aRow1 = aRow0 + (size_t)16 * K;
  const bf16* wTile = W + (size_t)nBase * K;

  // prime the pipeline: DMA first B tile
  if (wave == 0)
    tdm_load_tile64x32((unsigned)(uintptr_t)&Bs[0][0], wTile, (unsigned)K);

  int buf = 0;
  for (int kk = 0; kk < K; kk += 32, buf ^= 1) {
    const bool hasNext = (kk + 32) < K;
    if (wave == 0) {
      if (hasNext) {
        tdm_load_tile64x32((unsigned)(uintptr_t)&Bs[buf ^ 1][0],
                           wTile + kk + 32, (unsigned)K);
        __builtin_amdgcn_s_wait_tensorcnt(1);   // current tile complete
      } else {
        __builtin_amdgcn_s_wait_tensorcnt(0);
      }
    }
    __syncthreads();                            // B tile visible to all waves

    // A fragments (16x32): elems0-7 @ K=half*8, elems8-15 @ K=16+half*8
    const bf16* ap0 = aRow0 + kk + half * 8;
    const bf16* ap1 = aRow1 + kk + half * 8;
    if (hasNext) __builtin_prefetch(aRow0 + kk + 32, 0, 1);
    v16bf af0 = frag_cat(*(const v8bf*)ap0, *(const v8bf*)(ap0 + 16));
    v16bf af1 = frag_cat(*(const v8bf*)ap1, *(const v8bf*)(ap1 + 16));

#pragma unroll
    for (int t = 0; t < 4; ++t) {
      const bf16* bp = &Bs[buf][(t * 16 + l15) * 32 + half * 8];
      v16bf bfv = frag_cat(*(const v8bf*)bp, *(const v8bf*)(bp + 16));
      acc[0][t] = WMMA_BF16(af0, bfv, acc[0][t]);
      acc[1][t] = WMMA_BF16(af1, bfv, acc[1][t]);
    }
    __syncthreads();                            // reads done before buffer reuse
  }

  // ---- epilogues ----
#pragma unroll
  for (int p = 0; p < 2; ++p) {
#pragma unroll
    for (int t = 0; t < 4; ++t) {
      const int col = nBase + t * 16 + l15;
      const float bcol = bias[col];
#pragma unroll
      for (int r = 0; r < 8; ++r) {
        const int row = mBase + wave * 32 + p * 16 + r + half * 8;
        float val = acc[p][t][r] + bcol;

        if (EPI == EPI_QKV) {
          // scatter fused QKV output into per-(window,head) q/k/v tensors
          bf16* qkv = (bf16*)outp;
          const size_t SEG = (size_t)M * C;
          const int which = col / C;             // 0=q,1=k,2=v
          const int rem   = col - which * C;
          const int head  = rem >> 5;
          const int d     = rem & 31;
          const int win = row / NTOK, tok = row - (row / NTOK) * NTOK;
          if (which == 0) val *= SCALE;
          qkv[(size_t)which * SEG +
              ((size_t)(win * HEADS + head) * NTOK + tok) * HD + d] = (bf16)val;
        } else if (EPI == EPI_PROJ) {
          // window reverse + reverse cyclic shift + residual add (fp32)
          float* xres = (float*)outp;
          const int bi  = row / (NWIN * NTOK);
          const int r2  = row - bi * NWIN * NTOK;
          const int win = r2 / NTOK, tok = r2 - (r2 / NTOK) * NTOK;
          const int wh = win >> 3, ww = win & 7;
          const int th = tok / WSZ, tw = tok - (tok / WSZ) * WSZ;
          const int h = (wh * WSZ + th + SHIFT) % HH;    // roll(+SHIFT)
          const int w = (ww * WSZ + tw + SHIFT) % HH;
          const size_t idx = ((size_t)(bi * HH * HH + h * HH + w)) * C + col;
          xres[idx] = resid[idx] + val;
        } else if (EPI == EPI_FC1) {
          // exact GELU (erf), store bf16 hidden activations
          bf16* h2 = (bf16*)outp;
          const float gl = 0.5f * val * (1.f + erff(val * 0.70710678118654752f));
          h2[(size_t)row * N + col] = (bf16)gl;
        } else { // EPI_FC2
          float* op = (float*)outp;
          const size_t idx = (size_t)row * N + col;
          op[idx] = resid[idx] + val;
        }
      }
    }
  }
}

// ---------------------------------------------------------------------------
// Window attention: one block per (window, head).  N=49 padded to 64.
//   S = (Q*scale) K^T  (WMMA, K=32)  -> +rel-pos bias +shift mask
//   fp32 softmax in LDS -> P (bf16)  -> O = P V (WMMA, K=64, V^T in LDS)
// ---------------------------------------------------------------------------
__global__ void attn_kernel(const bf16* __restrict__ qb, const bf16* __restrict__ kb,
                            const bf16* __restrict__ vb, const float* __restrict__ rpb,
                            const float* __restrict__ mask, bf16* __restrict__ outp) {
  __shared__ __align__(16) bf16 Qs[64 * 32];
  __shared__ __align__(16) bf16 Ks[64 * 32];
  __shared__ __align__(16) bf16 VT[32 * 64];
  __shared__ __align__(16) float Sf[64 * 64];
  __shared__ __align__(16) bf16 Pb[64 * 64];

  const int wh   = blockIdx.x;                 // win*HEADS + head
  const int win  = wh / HEADS;
  const int head = wh - win * HEADS;
  const int tid  = threadIdx.x;
  const int wave = tid >> 5;
  const int lane = tid & 31;
  const int half = lane >> 4;
  const int l15  = lane & 15;

  // ---- load Q,K and transposed V (zero-pad tokens 49..63) ----
  const size_t base = (size_t)wh * NTOK * HD;
  for (int i = tid; i < 64 * 32; i += 128) {
    const int tok = i >> 5, d = i & 31;
    bf16 q = (bf16)0.f, k = (bf16)0.f, v = (bf16)0.f;
    if (tok < NTOK) {
      q = qb[base + tok * HD + d];
      k = kb[base + tok * HD + d];
      v = vb[base + tok * HD + d];
    }
    Qs[i] = q;
    Ks[i] = k;
    VT[d * 64 + tok] = v;
  }
  __syncthreads();

  // ---- S = Q K^T : wave handles rows wave*16..+15, single K=32 step ----
  v8f sacc[4];
#pragma unroll
  for (int t = 0; t < 4; ++t)
#pragma unroll
    for (int i = 0; i < 8; ++i) sacc[t][i] = 0.f;
  {
    const bf16* ap = Qs + (wave * 16 + l15) * 32 + half * 8;
    v16bf af = frag_cat(*(const v8bf*)ap, *(const v8bf*)(ap + 16));
#pragma unroll
    for (int t = 0; t < 4; ++t) {
      const bf16* bp = Ks + (t * 16 + l15) * 32 + half * 8;
      v16bf bfv = frag_cat(*(const v8bf*)bp, *(const v8bf*)(bp + 16));
      sacc[t] = WMMA_BF16(af, bfv, sacc[t]);
    }
  }

  // ---- add relative-position bias + shift mask, stage fp32 scores ----
  const float* mwin = mask + (size_t)(win & (NWIN - 1)) * NTOK * NTOK;
#pragma unroll
  for (int t = 0; t < 4; ++t) {
    const int m = t * 16 + l15;
#pragma unroll
    for (int r = 0; r < 8; ++r) {
      const int n = wave * 16 + r + half * 8;
      float s;
      if (m >= NTOK) {
        s = -1e30f;                              // padded keys
      } else if (n < NTOK) {
        const int n0 = n / WSZ, n1 = n - n0 * WSZ;
        const int m0 = m / WSZ, m1 = m - m0 * WSZ;
        const int ridx = (n0 - m0 + WSZ - 1) * (2 * WSZ - 1) + (n1 - m1 + WSZ - 1);
        s = sacc[t][r] + rpb[ridx * HEADS + head] + mwin[n * NTOK + m];
      } else {
        s = 0.f;                                 // padded queries: don't care
      }
      Sf[n * 64 + m] = s;
    }
  }
  __syncthreads();

  // ---- row softmax (one thread per row; 3 passes, no big locals) ----
  if (tid < 64) {
    const int n = tid;
    if (n < NTOK) {
      float mx = -1e30f;
      for (int m = 0; m < NTOK; ++m) mx = fmaxf(mx, Sf[n * 64 + m]);
      float sum = 0.f;
      for (int m = 0; m < NTOK; ++m) sum += expf(Sf[n * 64 + m] - mx);
      const float inv = 1.f / sum;
      for (int m = 0; m < NTOK; ++m) Pb[n * 64 + m] = (bf16)(expf(Sf[n * 64 + m] - mx) * inv);
      for (int m = NTOK; m < 64; ++m) Pb[n * 64 + m] = (bf16)0.f;
    } else {
      for (int m = 0; m < 64; ++m) Pb[n * 64 + m] = (bf16)0.f;
    }
  }
  __syncthreads();

  // ---- O = P V : K = 64 (two k-steps), N = 32 (two tiles) ----
  v8f oacc[2];
#pragma unroll
  for (int t = 0; t < 2; ++t)
#pragma unroll
    for (int i = 0; i < 8; ++i) oacc[t][i] = 0.f;
#pragma unroll
  for (int kk = 0; kk < 64; kk += 32) {
    const bf16* ap = Pb + (wave * 16 + l15) * 64 + kk + half * 8;
    v16bf af = frag_cat(*(const v8bf*)ap, *(const v8bf*)(ap + 16));
#pragma unroll
    for (int t = 0; t < 2; ++t) {
      const bf16* bp = VT + (t * 16 + l15) * 64 + kk + half * 8;
      v16bf bfv = frag_cat(*(const v8bf*)bp, *(const v8bf*)(bp + 16));
      oacc[t] = WMMA_BF16(af, bfv, oacc[t]);
    }
  }

  // ---- write attention output rows (win*49+n, head*32+d) as bf16 ----
#pragma unroll
  for (int t = 0; t < 2; ++t) {
    const int d = t * 16 + l15;
#pragma unroll
    for (int r = 0; r < 8; ++r) {
      const int n = wave * 16 + r + half * 8;
      if (n < NTOK)
        outp[((size_t)(win * NTOK + n)) * C + head * HD + d] = (bf16)oacc[t][r];
    }
  }
}

// ---------------------------------------------------------------------------
// Host-side launcher
// ---------------------------------------------------------------------------
extern "C" void kernel_launch(void* const* d_in, const int* in_sizes, int n_in,
                              void* d_out, int out_size, void* d_ws, size_t ws_size,
                              hipStream_t stream) {
  (void)in_sizes; (void)n_in; (void)out_size; (void)ws_size;

  const float* x      = (const float*)d_in[0];
  const float* maskm  = (const float*)d_in[1];
  const float* n1g    = (const float*)d_in[2];
  const float* n1b    = (const float*)d_in[3];
  const float* qkv_w  = (const float*)d_in[4];
  const float* qkv_b  = (const float*)d_in[5];
  const float* rpb    = (const float*)d_in[6];
  const float* proj_w = (const float*)d_in[7];
  const float* proj_b = (const float*)d_in[8];
  const float* n2g    = (const float*)d_in[9];
  const float* n2b    = (const float*)d_in[10];
  const float* fc1_w  = (const float*)d_in[11];
  const float* fc1_b  = (const float*)d_in[12];
  const float* fc2_w  = (const float*)d_in[13];
  const float* fc2_b  = (const float*)d_in[14];

  // ---- workspace layout (all offsets multiples of 256 bytes) ----
  const size_t SEG       = (size_t)TOKENS * C;          // 19,267,584 elements
  const size_t LN_BYTES  = SEG * sizeof(bf16);
  const size_t XR_BYTES  = SEG * sizeof(float);
  char* p = (char*)d_ws;
  bf16*  ln_buf = (bf16*)p;            p += LN_BYTES;
  float* xres   = (float*)p;           p += XR_BYTES;
  bf16*  wq = (bf16*)p;                p += (size_t)(3 * C) * C * sizeof(bf16);
  bf16*  wp = (bf16*)p;                p += (size_t)C * C * sizeof(bf16);
  bf16*  w1 = (bf16*)p;                p += (size_t)HID * C * sizeof(bf16);
  bf16*  w2 = (bf16*)p;                p += (size_t)C * HID * sizeof(bf16);
  bf16*  big = (bf16*)p;               // phase A: q|k|v|attn_out ; phase B: h2
  bf16*  qbuf = big;
  bf16*  kbuf = big + SEG;
  bf16*  vbuf = big + 2 * SEG;
  bf16*  attn_out = big + 3 * SEG;
  bf16*  h2 = big;                     // [TOKENS, HID] — overlaps dead q/k/v/attn

  // 1) weights fp32 -> bf16
  cvt_kernel<<<(3 * C * C + 255) / 256, 256, 0, stream>>>(qkv_w, wq, 3 * C * C);
  cvt_kernel<<<(C * C + 255) / 256, 256, 0, stream>>>(proj_w, wp, C * C);
  cvt_kernel<<<(HID * C + 255) / 256, 256, 0, stream>>>(fc1_w, w1, HID * C);
  cvt_kernel<<<(C * HID + 255) / 256, 256, 0, stream>>>(fc2_w, w2, C * HID);

  // 2) LN1 + cyclic shift + window partition (bf16, windowed row order)
  ln_kernel<<<TOKENS / 8, 256, 0, stream>>>(x, n1g, n1b, ln_buf, 1);

  // 3) fused QKV GEMM: [50176,384] x [1152,384]^T, scatter to q/k/v
  gemm_wmma<EPI_QKV><<<dim3((3 * C) / 64, TOKENS / 128), 128, 0, stream>>>(
      ln_buf, wq, qkv_b, (void*)qbuf, nullptr, TOKENS, 3 * C, C);

  // 4) window attention: 1024 windows x 12 heads
  attn_kernel<<<BATCH * NWIN * HEADS, 128, 0, stream>>>(
      qbuf, kbuf, vbuf, rpb, maskm, attn_out);

  // 5) proj GEMM + window reverse + unshift + residual -> xres (fp32)
  gemm_wmma<EPI_PROJ><<<dim3(C / 64, TOKENS / 128), 128, 0, stream>>>(
      attn_out, wp, proj_b, (void*)xres, x, TOKENS, C, C);

  // 6) LN2 (spatial order) -> ln_buf
  ln_kernel<<<TOKENS / 8, 256, 0, stream>>>(xres, n2g, n2b, ln_buf, 0);

  // 7) fc1 GEMM + exact GELU -> h2 (bf16)
  gemm_wmma<EPI_FC1><<<dim3(HID / 64, TOKENS / 128), 128, 0, stream>>>(
      ln_buf, w1, fc1_b, (void*)h2, nullptr, TOKENS, HID, C);

  // 8) fc2 GEMM + residual -> d_out (fp32)
  gemm_wmma<EPI_FC2><<<dim3(C / 64, TOKENS / 128), 128, 0, stream>>>(
      h2, w2, fc2_b, d_out, xres, TOKENS, C, HID);
}
